// ZendoNet_13134009991819
// MI455X (gfx1250) — compile-verified, matrix operand order checked by reading, not run on hardware
//
#include <hip/hip_runtime.h>
#include <cstdint>

#define NN 65536       // N_NODES
#define NE 1048576     // N_EDGES
#define NG 512         // N_GRAPHS
#define FIN 16
#define HID 64
#define BN_EPS 1e-5f
#define L2_EPS 1e-12f

typedef __attribute__((ext_vector_type(16))) _Float16 v16h;
typedef __attribute__((ext_vector_type(8)))  _Float16 v8h;
typedef __attribute__((ext_vector_type(8)))  float    v8f;

// ---------------------------------------------------------------- fill
__global__ void fill_zero(float* __restrict__ p, long long n) {
  long long i = blockIdx.x * (long long)blockDim.x + threadIdx.x;
  if (i < n) p[i] = 0.f;
}

// ------------------------------------------------- edge gather/scatter-add
// aggr[dst[e], :] += x[src[e], :]   (one thread per edge x 4-float chunk)
__global__ void scatter_add(const float* __restrict__ x, const int* __restrict__ src,
                            const int* __restrict__ dst, float* __restrict__ aggr,
                            int E, int F) {
  long long t = blockIdx.x * (long long)blockDim.x + threadIdx.x;
  int chunks = F >> 2;
  long long total = (long long)E * chunks;
  if (t >= total) return;
  int e = (int)(t / chunks);
  int c = (int)(t % chunks) << 2;
  int s = src[e], d = dst[e];
  const float4 v = *(const float4*)(x + (size_t)s * F + c);
  float* p = aggr + (size_t)d * F + c;
  atomicAdd(p + 0, v.x);
  atomicAdd(p + 1, v.y);
  atomicAdd(p + 2, v.z);
  atomicAdd(p + 3, v.w);
}

// ------------------------------------------------------------ WMMA GEMM
// C[M x 64] = (A[M x K] (+ Aadd)) @ W[K x 64] + bias   with K in {16, 64}
// 256 threads = 8 waves; each wave: one 16-row stripe, all 4 N-tiles.
// W is pre-transposed into LDS as f16 so every B fragment is two
// contiguous 16B runs (ds_load_b128 x2); every A fragment is two
// contiguous 32B runs (global_load_b128 x2 [+x2 for the fused add]).
template <int K, bool HASADD>
__global__ void gemm_wmma(const float* __restrict__ A, const float* __restrict__ Aadd,
                          const float* __restrict__ W, const float* __restrict__ bias,
                          float* __restrict__ C) {
  __shared__ __align__(16) _Float16 WT[64 * 64];   // [n][k], zero-padded K
  for (int i = threadIdx.x; i < 64 * 64; i += 256) {
    int n = i >> 6, k = i & 63;
    WT[i] = (k < K) ? (_Float16)W[(size_t)k * 64 + n] : (_Float16)0.f;
  }
  __syncthreads();

  const int lane = threadIdx.x & 31;
  const int wave = threadIdx.x >> 5;
  const int row0 = (blockIdx.x * 8 + wave) * 16;   // exact: grid = M/128
  const int g  = lane >> 4;       // lane group
  const int mn = lane & 15;       // A row within tile / B,C column

  const float* rowp  = A + (size_t)(row0 + mn) * K;
  const float* rowp2 = HASADD ? (Aadd + (size_t)(row0 + mn) * K) : nullptr;

  v8f acc[4] = {};
#pragma unroll
  for (int k0 = 0; k0 < K; k0 += 32) {
    v16h a;
    // halves [0..7]: k = k0 + 8g .. +7  (always in range: k0+16 <= K)
    {
      const float4* p = (const float4*)(rowp + k0 + 8 * g);
      float4 q0 = p[0], q1 = p[1];
      if (HASADD) {
        const float4* p2 = (const float4*)(rowp2 + k0 + 8 * g);
        float4 r0 = p2[0], r1 = p2[1];
        q0.x += r0.x; q0.y += r0.y; q0.z += r0.z; q0.w += r0.w;
        q1.x += r1.x; q1.y += r1.y; q1.z += r1.z; q1.w += r1.w;
      }
      a[0] = (_Float16)q0.x; a[1] = (_Float16)q0.y;
      a[2] = (_Float16)q0.z; a[3] = (_Float16)q0.w;
      a[4] = (_Float16)q1.x; a[5] = (_Float16)q1.y;
      a[6] = (_Float16)q1.z; a[7] = (_Float16)q1.w;
    }
    // halves [8..15]: k = k0 + 16 + 8g .. +7 (zero-pad when K==16)
    if (k0 + 32 <= K) {
      const float4* p = (const float4*)(rowp + k0 + 16 + 8 * g);
      float4 q0 = p[0], q1 = p[1];
      if (HASADD) {
        const float4* p2 = (const float4*)(rowp2 + k0 + 16 + 8 * g);
        float4 r0 = p2[0], r1 = p2[1];
        q0.x += r0.x; q0.y += r0.y; q0.z += r0.z; q0.w += r0.w;
        q1.x += r1.x; q1.y += r1.y; q1.z += r1.z; q1.w += r1.w;
      }
      a[8]  = (_Float16)q0.x; a[9]  = (_Float16)q0.y;
      a[10] = (_Float16)q0.z; a[11] = (_Float16)q0.w;
      a[12] = (_Float16)q1.x; a[13] = (_Float16)q1.y;
      a[14] = (_Float16)q1.z; a[15] = (_Float16)q1.w;
    } else {
#pragma unroll
      for (int i = 8; i < 16; ++i) a[i] = (_Float16)0.f;
    }

#pragma unroll
    for (int nt = 0; nt < 4; ++nt) {
      const _Float16* wp = &WT[(size_t)(nt * 16 + mn) * 64 + k0 + 8 * g];
      v8h b0 = *(const v8h*)(wp);        // k0+8g    .. +7
      v8h b1 = *(const v8h*)(wp + 16);   // k0+16+8g .. +7
      v16h b;
#pragma unroll
      for (int i = 0; i < 8; ++i) { b[i] = b0[i]; b[8 + i] = b1[i]; }
      acc[nt] = __builtin_amdgcn_wmma_f32_16x16x32_f16(
          false, a, false, b, (short)0, acc[nt], false, false);
    }
  }

  // D layout: reg r -> row row0 + r + 8g, col = nt*16 + mn
#pragma unroll
  for (int nt = 0; nt < 4; ++nt) {
    float bv = bias[nt * 16 + mn];
#pragma unroll
    for (int r = 0; r < 8; ++r)
      C[(size_t)(row0 + r + 8 * g) * 64 + nt * 16 + mn] = acc[nt][r] + bv;
  }
}

// ------------------------------------------------- batch-norm statistics
__global__ void col_stats(const float* __restrict__ H, float* __restrict__ sum,
                          float* __restrict__ sumsq, int M) {
  int col = threadIdx.x & 63;
  int sl  = (threadIdx.x >> 6) + (blockIdx.x << 2);
  int nsl = gridDim.x << 2;
  float s = 0.f, s2 = 0.f;
  for (int r = sl; r < M; r += nsl) {
    float v = H[(size_t)r * 64 + col];
    s += v; s2 += v * v;
  }
  atomicAdd(&sum[col], s);
  atomicAdd(&sumsq[col], s2);
}

// -------------------------------------------- BN(train stats)+ReLU (+res)
__global__ void bn_relu(const float* __restrict__ H, float* __restrict__ out,
                        const float* __restrict__ sum, const float* __restrict__ sumsq,
                        const float* __restrict__ gamma, const float* __restrict__ beta,
                        const float* __restrict__ residual, int M) {
  long long i = blockIdx.x * (long long)blockDim.x + threadIdx.x;
  if (i >= (long long)M * 64) return;
  int col = (int)(i & 63);
  float inv = 1.f / (float)M;
  float mean = sum[col] * inv;
  float var  = sumsq[col] * inv - mean * mean;   // biased variance
  float v = (H[i] - mean) * rsqrtf(var + BN_EPS) * gamma[col] + beta[col];
  v = fmaxf(v, 0.f);
  out[i] = residual ? residual[i] + v : v;
}

// ------------------------------------------------------- global_add_pool
__global__ void pool_kernel(const float* __restrict__ x, const int* __restrict__ batch,
                            float* __restrict__ gout) {
  long long i = blockIdx.x * (long long)blockDim.x + threadIdx.x;
  if (i >= (long long)NN * HID) return;
  int node = (int)(i >> 6);
  int c    = (int)(i & 63);
  atomicAdd(&gout[(size_t)batch[node] * HID + c], x[i]);
}

// ------------------------------------------- projection head + L2 norm
__global__ void head_kernel(const float* __restrict__ g, const float* __restrict__ W1,
                            const float* __restrict__ b1, const float* __restrict__ W2,
                            const float* __restrict__ b2, float* __restrict__ out,
                            int dh, int dout) {
  __shared__ float hbuf[64];
  __shared__ float zbuf[64];
  __shared__ float snorm;
  int row = blockIdx.x;
  int t = threadIdx.x;
  if (t < dh) {
    float acc = b1[t];
    for (int k = 0; k < HID; ++k) acc += g[(size_t)row * HID + k] * W1[k * dh + t];
    hbuf[t] = fmaxf(acc, 0.f);
  }
  __syncthreads();
  if (t < dout) {
    float acc = b2[t];
    for (int k = 0; k < dh; ++k) acc += hbuf[k] * W2[k * dout + t];
    zbuf[t] = acc;
  }
  __syncthreads();
  if (t == 0) {
    float s = 0.f;
    for (int k = 0; k < dout; ++k) s += zbuf[k] * zbuf[k];
    snorm = fmaxf(sqrtf(s), L2_EPS);
  }
  __syncthreads();
  if (t < dout) out[(size_t)row * dout + t] = zbuf[t] / snorm;
}

// =====================================================================
extern "C" void kernel_launch(void* const* d_in, const int* in_sizes, int n_in,
                              void* d_out, int out_size, void* d_ws, size_t ws_size,
                              hipStream_t stream) {
  (void)in_sizes; (void)n_in; (void)out_size; (void)ws_size;
  const float* x_in  = (const float*)d_in[0];
  const int*   src   = (const int*)d_in[1];
  const int*   dst   = src + NE;
  const int*   batch = (const int*)d_in[2];

  // params: conv{1,2,3} at 3,11,19 (W1,b1,g1,be1,W2,b2,g2,be2);
  // heads color/size/ground/struct at 27,31,35,39 (W1,b1,W2,b2)
  auto F = [&](int i) { return (const float*)d_in[i]; };

  // workspace carve
  float* R0    = (float*)d_ws;                    // current node features [NN x 64]
  float* R1    = R0 + (size_t)NN * HID;           // aggr buffer
  float* R2    = R1 + (size_t)NN * HID;           // h1
  float* R3    = R2 + (size_t)NN * HID;           // h2
  float* ssum  = R3 + (size_t)NN * HID;           // [64]
  float* ssq   = ssum + HID;                      // [64]
  float* gpool = ssq + HID;                       // [NG x 64]

  auto fill = [&](float* p, long long n) {
    fill_zero<<<(unsigned)((n + 255) / 256), 256, 0, stream>>>(p, n);
  };

  auto run_gin = [&](const float* xin, int K, int base, float* out_p,
                     const float* residual) {
    // aggr = segment_sum(x[src], dst)
    fill(R1, (long long)NN * K);
    long long st = (long long)NE * (K / 4);
    scatter_add<<<(unsigned)((st + 255) / 256), 256, 0, stream>>>(xin, src, dst, R1, NE, K);
    // h1 = (x + aggr) @ W1 + b1 ; BN+ReLU
    if (K == 16)
      gemm_wmma<16, true><<<NN / 128, 256, 0, stream>>>(xin, R1, F(base + 0),
                                                        F(base + 1), R2);
    else
      gemm_wmma<64, true><<<NN / 128, 256, 0, stream>>>(xin, R1, F(base + 0),
                                                        F(base + 1), R2);
    fill(ssum, 2 * HID);
    col_stats<<<256, 256, 0, stream>>>(R2, ssum, ssq, NN);
    bn_relu<<<(NN * HID) / 256, 256, 0, stream>>>(R2, R2, ssum, ssq, F(base + 2),
                                                  F(base + 3), nullptr, NN);
    // h2 = h1 @ W2 + b2 ; BN+ReLU (+ residual)
    gemm_wmma<64, false><<<NN / 128, 256, 0, stream>>>(R2, nullptr, F(base + 4),
                                                       F(base + 5), R3);
    fill(ssum, 2 * HID);
    col_stats<<<256, 256, 0, stream>>>(R3, ssum, ssq, NN);
    bn_relu<<<(NN * HID) / 256, 256, 0, stream>>>(R3, out_p, ssum, ssq, F(base + 6),
                                                  F(base + 7), residual, NN);
  };

  run_gin(x_in, FIN, 3, R0, nullptr);   // x1 = gin1(x)
  run_gin(R0, HID, 11, R0, R0);         // x2 = x1 + gin2(x1)
  run_gin(R0, HID, 19, R0, R0);         // x3 = x2 + gin3(x2)

  // global_add_pool
  fill(gpool, (long long)NG * HID);
  pool_kernel<<<(NN * HID) / 256, 256, 0, stream>>>(R0, batch, gpool);

  // heads -> d_out: color[512x16], size[512x16], ground[512x8], struct[512x32]
  float* out = (float*)d_out;
  head_kernel<<<NG, 64, 0, stream>>>(gpool, F(27), F(28), F(29), F(30),
                                     out, 32, 16);
  head_kernel<<<NG, 64, 0, stream>>>(gpool, F(31), F(32), F(33), F(34),
                                     out + (size_t)NG * 16, 32, 16);
  head_kernel<<<NG, 64, 0, stream>>>(gpool, F(35), F(36), F(37), F(38),
                                     out + (size_t)NG * 32, 16, 8);
  head_kernel<<<NG, 64, 0, stream>>>(gpool, F(39), F(40), F(41), F(42),
                                     out + (size_t)NG * 40, 64, 32);
}
